// TransformerModel_44212393345817
// MI455X (gfx1250) — matrix operand order repaired and checked by reading, not compiled, not run on hardware
//
#include <hip/hip_runtime.h>
#include <hip/hip_bf16.h>
#include <math.h>

// ---------------------------------------------------------------------------
// Fused transformer encoder layer for MI455X (gfx1250, wave32, WMMA).
//   B=8192, S=32, HID=64, NH=8, D=8, FFN=64, E=131072, EMB=64
// Strategy: compact the sparse edge bias into a 32x32x8 table T (32KB),
// then one fully fused block per sequence using v_wmma_f32_16x16x32_f16
// for all six 32x64 @ 64x64 GEMMs. Memory traffic ~170MB -> HBM-bound.
// ---------------------------------------------------------------------------

typedef _Float16 half_t;
typedef __attribute__((ext_vector_type(16))) _Float16 v16h;
typedef __attribute__((ext_vector_type(8)))  _Float16 v8h;
typedef __attribute__((ext_vector_type(8)))  float    v8f;

#define SS   32
#define HIDN 64
#define NHD  8
#define DHD  8
#define EMBD 64
#define LDH  72          // padded f16 row stride (144B, 16B aligned)
#define SCALE 0.35355339059327379f   // 8^-0.5
#define LN_EPS 1e-5f

// ---- WMMA fragment loaders (CDNA5 16-bit layouts, CDNA5 ISA) ----
// A (16x32 f16): lane L holds row M=L&15; lanes 0-15 cover K in [0,8)+[16,24),
// lanes 16-31 cover K in [8,16)+[24,32). Each half-group is 16 contiguous B.
static __device__ __forceinline__ v16h load_frag_a(const half_t* base, int ld,
                                                   int m0, int k0) {
  int lane = threadIdx.x & 31;
  int m  = m0 + (lane & 15);
  int kg = (lane >> 4) * 8;
  const half_t* p = base + m * ld + k0 + kg;
  v8h lo = *(const v8h*)(p);        // K = k0+kg .. +8
  v8h hi = *(const v8h*)(p + 16);   // K = k0+16+kg .. +8
  v16h r;
#pragma unroll
  for (int i = 0; i < 8; ++i) { r[i] = lo[i]; r[i + 8] = hi[i]; }
  return r;
}

// B (32x16 f16): lane L holds column N=L&15; lanes 0-15 hold K in [0,8)+[16,24),
// lanes 16-31 hold K in [8,16)+[24,32). Weights stored transposed [n][k] so
// each 8-K group per lane is contiguous -> two 16B global loads.
static __device__ __forceinline__ v16h load_frag_b(const half_t* wt,
                                                   int n0, int k0) {
  int lane = threadIdx.x & 31;
  int n  = n0 + (lane & 15);
  int kg = (lane >> 4) * 8;
  const half_t* p = wt + n * HIDN + k0 + kg;
  v8h lo = *(const v8h*)(p);
  v8h hi = *(const v8h*)(p + 16);
  v16h r;
#pragma unroll
  for (int i = 0; i < 8; ++i) { r[i] = lo[i]; r[i + 8] = hi[i]; }
  return r;
}

// One 16x16 output tile of a [32,64]x[64,64] GEMM: two K=32 WMMA steps.
static __device__ __forceinline__ v8f wmma_tile(const half_t* A, int lda,
                                                const half_t* Bt,
                                                int m0, int n0) {
  v8f c = {};
#pragma unroll
  for (int k0 = 0; k0 < HIDN; k0 += 32) {
    v16h a = load_frag_a(A, lda, m0, k0);
    v16h b = load_frag_b(Bt, n0, k0);
    c = __builtin_amdgcn_wmma_f32_16x16x32_f16(false, a, false, b,
                                               (short)0, c, false, false);
  }
  return c;
}

// C/D f32 layout: lane holds N=lane&15; VGPR r -> M = r + (lane>=16 ? 8 : 0).
static __device__ __forceinline__ void store_tile_f16_bias(
    half_t* dst, int ld, int m0, int n0, v8f c, const float* __restrict__ bias) {
  int lane = threadIdx.x & 31;
  int n  = n0 + (lane & 15);
  int mb = m0 + ((lane & 16) ? 8 : 0);
  float bv = bias[n];
#pragma unroll
  for (int r = 0; r < 8; ++r)
    dst[(mb + r) * ld + n] = (half_t)(c[r] + bv);
}

// ---------------------------------------------------------------------------
// Kernel 1: zero the bias table T[32][32][8] and build transposed f16 weights.
// ---------------------------------------------------------------------------
__global__ void prep_kernel(const float* __restrict__ wq, const float* __restrict__ wk,
                            const float* __restrict__ wv, const float* __restrict__ wo,
                            const float* __restrict__ w1, const float* __restrict__ w2,
                            float* __restrict__ T, half_t* __restrict__ wts) {
  int i = blockIdx.x * blockDim.x + threadIdx.x;
  if (i < SS * SS * NHD) T[i] = 0.0f;
  if (i < 6 * HIDN * HIDN) {
    int m   = i >> 12;            // which matrix
    int idx = i & 4095;
    int k = idx >> 6, n = idx & 63;
    const float* W;
    switch (m) {
      case 0: W = wq; break; case 1: W = wk; break; case 2: W = wv; break;
      case 3: W = wo; break; case 4: W = w1; break; default: W = w2; break;
    }
    wts[m * HIDN * HIDN + n * HIDN + k] = (half_t)W[k * HIDN + n];
  }
}

// ---------------------------------------------------------------------------
// Kernel 2: T[s][t][h] += edge_attr[e,:] @ we[:,h] + be[h] for each edge.
// ---------------------------------------------------------------------------
__global__ __launch_bounds__(256)
void edge_bias_kernel(const float* __restrict__ ea, const int* __restrict__ eidx,
                      const float* __restrict__ we, const float* __restrict__ be,
                      float* __restrict__ T, int E) {
  __shared__ float wes[EMBD * NHD];
  __shared__ float bes[NHD];
  for (int i = threadIdx.x; i < EMBD * NHD; i += blockDim.x) wes[i] = we[i];
  if (threadIdx.x < NHD) bes[threadIdx.x] = be[threadIdx.x];
  __syncthreads();

  int e = blockIdx.x * blockDim.x + threadIdx.x;
  if (e >= E) return;
  int src = eidx[e];
  int dst = eidx[E + e];
  float acc[NHD];
#pragma unroll
  for (int h = 0; h < NHD; ++h) acc[h] = bes[h];
  const float* row = ea + (size_t)e * EMBD;
  for (int k = 0; k < EMBD; k += 4) {
    float4 a4 = *(const float4*)(row + k);
#pragma unroll
    for (int h = 0; h < NHD; ++h) {
      acc[h] += a4.x * wes[(k + 0) * NHD + h];
      acc[h] += a4.y * wes[(k + 1) * NHD + h];
      acc[h] += a4.z * wes[(k + 2) * NHD + h];
      acc[h] += a4.w * wes[(k + 3) * NHD + h];
    }
  }
  int base = (src * SS + dst) * NHD;
#pragma unroll
  for (int h = 0; h < NHD; ++h) atomicAdd(&T[base + h], acc[h]);
}

// ---------------------------------------------------------------------------
// Kernel 3: fully fused encoder layer, one 4-wave block per sequence.
// ---------------------------------------------------------------------------
__global__ __launch_bounds__(128)
void fused_layer_kernel(const float* __restrict__ x, const int* __restrict__ batch,
                        const float* __restrict__ bq, const float* __restrict__ bk,
                        const float* __restrict__ bv, const float* __restrict__ bo,
                        const float* __restrict__ b1, const float* __restrict__ b2,
                        const float* __restrict__ ln1g, const float* __restrict__ ln1b,
                        const float* __restrict__ ln2g, const float* __restrict__ ln2b,
                        const float* __restrict__ Tg, const half_t* __restrict__ wts,
                        float* __restrict__ out) {
  __shared__ float  xs[SS * HIDN];                 // running residual (f32)
  __shared__ half_t ybuf[SS * LDH];                // LN output (A operand)
  __shared__ half_t qbuf[SS * LDH];
  __shared__ half_t kbuf[SS * LDH];
  __shared__ half_t vbuf[SS * LDH];
  __shared__ half_t aobuf[SS * LDH];               // attention output (concat heads)
  __shared__ half_t hbuf[SS * LDH];                // GELU output

  const int b    = blockIdx.x;
  const int tid  = threadIdx.x;
  const int wave = tid >> 5;
  const half_t* wq_t = wts + 0 * HIDN * HIDN;
  const half_t* wk_t = wts + 1 * HIDN * HIDN;
  const half_t* wv_t = wts + 2 * HIDN * HIDN;
  const half_t* wo_t = wts + 3 * HIDN * HIDN;
  const half_t* w1_t = wts + 4 * HIDN * HIDN;
  const half_t* w2_t = wts + 5 * HIDN * HIDN;

  // ---- load x[b] : 2048 floats via float4 ----
  const float* xg = x + (size_t)b * SS * HIDN;
#pragma unroll
  for (int i = 0; i < 4; ++i)
    ((float4*)xs)[tid + i * 128] = ((const float4*)xg)[tid + i * 128];
  __syncthreads();

  // ---- LayerNorm 1 -> ybuf (f16) ----
  if (tid < SS) {
    const float* row = xs + tid * HIDN;
    float m = 0.f;
    for (int c = 0; c < HIDN; ++c) m += row[c];
    m *= (1.0f / HIDN);
    float v = 0.f;
    for (int c = 0; c < HIDN; ++c) { float d = row[c] - m; v += d * d; }
    v *= (1.0f / HIDN);
    float inv = rsqrtf(v + LN_EPS);
    for (int c = 0; c < HIDN; ++c)
      ybuf[tid * LDH + c] = (half_t)((row[c] - m) * inv * ln1g[c] + ln1b[c]);
  }
  __syncthreads();

  // ---- Q, K, V projections (WMMA): 8 tiles each, 2 per wave ----
  for (int t = wave; t < 8; t += 4) {
    int mt = (t >> 2) * 16, nt = (t & 3) * 16;
    store_tile_f16_bias(qbuf, LDH, mt, nt, wmma_tile(ybuf, LDH, wq_t, mt, nt), bq);
    store_tile_f16_bias(kbuf, LDH, mt, nt, wmma_tile(ybuf, LDH, wk_t, mt, nt), bk);
    store_tile_f16_bias(vbuf, LDH, mt, nt, wmma_tile(ybuf, LDH, wv_t, mt, nt), bv);
  }
  __syncthreads();

  // ---- Attention: 256 (head,row) pairs over 128 threads (D=8 -> VALU) ----
#pragma unroll
  for (int pp = 0; pp < 2; ++pp) {
    int p = tid + pp * 128;
    int h = p >> 5, s = p & 31;
    float qv[DHD];
#pragma unroll
    for (int d = 0; d < DHD; ++d) qv[d] = (float)qbuf[s * LDH + h * DHD + d];
    bool has_bias = (batch[s] == b);
    const float* Trow = Tg + (s * SS) * NHD + h;
    float sc[SS];
    float mx = -3.4e38f;
    for (int t = 0; t < SS; ++t) {
      float dot = 0.f;
#pragma unroll
      for (int d = 0; d < DHD; ++d) dot += qv[d] * (float)kbuf[t * LDH + h * DHD + d];
      float bias = has_bias ? Trow[t * NHD] : 0.f;
      float val = dot * SCALE + bias;
      sc[t] = val;
      mx = fmaxf(mx, val);
    }
    float sum = 0.f;
    for (int t = 0; t < SS; ++t) { sc[t] = __expf(sc[t] - mx); sum += sc[t]; }
    float invs = 1.f / sum;
    float o[DHD];
#pragma unroll
    for (int d = 0; d < DHD; ++d) o[d] = 0.f;
    for (int t = 0; t < SS; ++t) {
      float a = sc[t] * invs;
#pragma unroll
      for (int d = 0; d < DHD; ++d) o[d] += a * (float)vbuf[t * LDH + h * DHD + d];
    }
#pragma unroll
    for (int d = 0; d < DHD; ++d) aobuf[s * LDH + h * DHD + d] = (half_t)o[d];
  }
  __syncthreads();

  // ---- Output projection + residual: xs += ao @ wo + bo ----
  for (int t = wave; t < 8; t += 4) {
    int mt = (t >> 2) * 16, nt = (t & 3) * 16;
    v8f c = wmma_tile(aobuf, LDH, wo_t, mt, nt);
    int lane = tid & 31;
    int n  = nt + (lane & 15);
    int mb = mt + ((lane & 16) ? 8 : 0);
    float bv_ = bo[n];
#pragma unroll
    for (int r = 0; r < 8; ++r) xs[(mb + r) * HIDN + n] += c[r] + bv_;
  }
  __syncthreads();

  // ---- LayerNorm 2 -> ybuf (reuse) ----
  if (tid < SS) {
    const float* row = xs + tid * HIDN;
    float m = 0.f;
    for (int c = 0; c < HIDN; ++c) m += row[c];
    m *= (1.0f / HIDN);
    float v = 0.f;
    for (int c = 0; c < HIDN; ++c) { float d = row[c] - m; v += d * d; }
    v *= (1.0f / HIDN);
    float inv = rsqrtf(v + LN_EPS);
    for (int c = 0; c < HIDN; ++c)
      ybuf[tid * LDH + c] = (half_t)((row[c] - m) * inv * ln2g[c] + ln2b[c]);
  }
  __syncthreads();

  // ---- FFN layer 1 + exact GELU -> hbuf ----
  for (int t = wave; t < 8; t += 4) {
    int mt = (t >> 2) * 16, nt = (t & 3) * 16;
    v8f c = wmma_tile(ybuf, LDH, w1_t, mt, nt);
    int lane = tid & 31;
    int n  = nt + (lane & 15);
    int mb = mt + ((lane & 16) ? 8 : 0);
    float bv_ = b1[n];
#pragma unroll
    for (int r = 0; r < 8; ++r) {
      float g = c[r] + bv_;
      float ge = 0.5f * g * (1.0f + erff(g * 0.70710678118654752f));
      hbuf[(mb + r) * LDH + n] = (half_t)ge;
    }
  }
  __syncthreads();

  // ---- FFN layer 2 + residual -> global out ----
  float* og = out + (size_t)b * SS * HIDN;
  for (int t = wave; t < 8; t += 4) {
    int mt = (t >> 2) * 16, nt = (t & 3) * 16;
    v8f c = wmma_tile(hbuf, LDH, w2_t, mt, nt);
    int lane = tid & 31;
    int n  = nt + (lane & 15);
    int mb = mt + ((lane & 16) ? 8 : 0);
    float bv_ = b2[n];
#pragma unroll
    for (int r = 0; r < 8; ++r)
      og[(mb + r) * HIDN + n] = xs[(mb + r) * HIDN + n] + c[r] + bv_;
  }
}

// ---------------------------------------------------------------------------
extern "C" void kernel_launch(void* const* d_in, const int* in_sizes, int n_in,
                              void* d_out, int out_size, void* d_ws, size_t ws_size,
                              hipStream_t stream) {
  const float* x         = (const float*)d_in[0];
  const float* edge_attr = (const float*)d_in[1];
  const int*   edge_idx  = (const int*)d_in[2];
  const int*   batch     = (const int*)d_in[3];
  const float* wq = (const float*)d_in[4];  const float* bq = (const float*)d_in[5];
  const float* wk = (const float*)d_in[6];  const float* bk = (const float*)d_in[7];
  const float* wv = (const float*)d_in[8];  const float* bv = (const float*)d_in[9];
  const float* we = (const float*)d_in[10]; const float* be = (const float*)d_in[11];
  const float* wo = (const float*)d_in[12]; const float* bo = (const float*)d_in[13];
  const float* ln1g = (const float*)d_in[14]; const float* ln1b = (const float*)d_in[15];
  const float* ln2g = (const float*)d_in[16]; const float* ln2b = (const float*)d_in[17];
  const float* w1 = (const float*)d_in[18]; const float* b1 = (const float*)d_in[19];
  const float* w2 = (const float*)d_in[20]; const float* b2 = (const float*)d_in[21];

  int Bn = in_sizes[0] / (SS * HIDN);   // 8192
  int E  = in_sizes[1] / EMBD;          // 131072

  float*  T   = (float*)d_ws;                                   // 32 KB
  half_t* wts = (half_t*)((char*)d_ws + SS * SS * NHD * sizeof(float)); // 48 KB

  prep_kernel<<<96, 256, 0, stream>>>(wq, wk, wv, wo, w1, w2, T, wts);
  edge_bias_kernel<<<(E + 255) / 256, 256, 0, stream>>>(edge_attr, edge_idx,
                                                        we, be, T, E);
  fused_layer_kernel<<<Bn, 128, 0, stream>>>(x, batch, bq, bk, bv, bo, b1, b2,
                                             ln1g, ln1b, ln2g, ln2b, T, wts,
                                             (float*)d_out);
}
